// MambaBlock_11424613007364
// MI455X (gfx1250) — matrix operand order repaired and checked by reading, not compiled, not run on hardware
//
#include <hip/hip_runtime.h>
#include <hip/hip_bf16.h>
#include <math.h>
#include <stdint.h>

// ---- problem dims (fixed by reference) ----
#define D_MODEL 1024
#define D_INNER 2048
#define D_STATE 16
#define DT_RANK 64
#define D_CONV  4
#define BB      2
#define LL      1024
#define M_TOK   (BB * LL)          // 2048 token rows

typedef __attribute__((ext_vector_type(16))) __bf16 v16bf;
typedef __attribute__((ext_vector_type(8)))  __bf16 v8bf;
typedef __attribute__((ext_vector_type(8)))  float  v8f;
typedef unsigned int  u32x4 __attribute__((ext_vector_type(4)));
typedef int           i32x8 __attribute__((ext_vector_type(8)));
typedef int           i32x4 __attribute__((ext_vector_type(4)));

__device__ __forceinline__ v16bf cat16(v8bf lo, v8bf hi) {
  return __builtin_shufflevector(lo, hi, 0,1,2,3,4,5,6,7,8,9,10,11,12,13,14,15);
}

__device__ __forceinline__ float softplus_f(float x) {
  return (x > 20.0f) ? x : log1pf(__expf(x));
}
__device__ __forceinline__ float silu_f(float x) {
  return x / (1.0f + __expf(-x));
}

// ---------------------------------------------------------------------------
// bf16 fragment loaders (layouts per CDNA5 ISA §7.12.2, wave32)
// ---------------------------------------------------------------------------
__device__ __forceinline__ v16bf load_A_frag_bf(const __bf16* __restrict__ A, int lda,
                                                int row0, int k0, int lane) {
  const int m  = lane & 15;
  const int kh = (lane >> 4) << 3;                 // 0 or 8
  const __bf16* p = A + (size_t)(row0 + m) * lda + k0;
  return cat16(*(const v8bf*)(p + kh), *(const v8bf*)(p + 16 + kh));
}

__device__ __forceinline__ v16bf load_B_frag_bf(const __bf16* __restrict__ W, int ldw,
                                                int col0, int k0, int lane) {
  const int n  = lane & 15;
  const int kb = k0 + ((lane >> 4) << 3);
  const __bf16* p = W + (size_t)(col0 + n) * ldw + kb;
  return cat16(*(const v8bf*)p, *(const v8bf*)(p + 16));
}

// ---------------------------------------------------------------------------
// TDM: 2D bf16 tile load (tile_d1 rows x tile_d0 elems) into LDS.
// D# packing per CDNA5 ISA.
// ---------------------------------------------------------------------------
__device__ __forceinline__ void tdm_load_2d_bf16(const __bf16* gptr, unsigned lds_off,
                                                 unsigned tensor_d0, unsigned tensor_d1,
                                                 unsigned tile_d0, unsigned tile_d1,
                                                 unsigned stride_d0) {
  unsigned long long ga = (unsigned long long)(uintptr_t)gptr;
  u32x4 g0;
  g0[0] = 1u;                                              // count=1 valid user D#
  g0[1] = lds_off;                                         // LDS byte address
  g0[2] = (unsigned)(ga & 0xFFFFFFFFu);                    // global_addr[31:0]
  g0[3] = (unsigned)((ga >> 32) & 0x1FFFFFFu) | (2u << 30);// addr[56:32] | type=2
  i32x8 g1;
  g1[0] = (1 << 16);                                       // data_size=1 (2B), mask=0
  g1[1] = (int)((tensor_d0 & 0xFFFFu) << 16);              // tensor_dim0[15:0] @bit48
  g1[2] = (int)(((tensor_d0 >> 16) & 0xFFFFu) |
                ((tensor_d1 & 0xFFFFu) << 16));            // dim0 hi | dim1 lo
  g1[3] = (int)(((tensor_d1 >> 16) & 0xFFFFu) |
                ((tile_d0 & 0xFFFFu) << 16));              // dim1 hi | tile_dim0
  g1[4] = (int)(tile_d1 & 0xFFFFu);                        // tile_dim1, tile_dim2=0
  g1[5] = (int)stride_d0;                                  // tensor_dim0_stride lo32
  g1[6] = 0;
  g1[7] = 0;
  i32x4 gz = {0, 0, 0, 0};
#if defined(__clang_major__) && (__clang_major__ >= 23)
  i32x8 gz8 = {0, 0, 0, 0, 0, 0, 0, 0};
  __builtin_amdgcn_tensor_load_to_lds(g0, g1, gz, gz, gz8, 0);
#else
  __builtin_amdgcn_tensor_load_to_lds(g0, g1, gz, gz, 0);
#endif
}

// ---------------------------------------------------------------------------
// Big GEMM: C[M,N] = act(A@W^T + bias).  Block = 8 waves = 64M x 64N tile.
// Both operand tiles (64x32 bf16) staged in LDS by the Tensor Data Mover,
// double-buffered so the DMA of stage i+1 overlaps compute of stage i.
// TENSORcnt is in-order per wave: after issuing 2 new loads on top of 2
// outstanding, s_wait_tensorcnt(2) guarantees the current stage is resident.
// Requires M%64==0, N%64==0, K%32==0.
// ---------------------------------------------------------------------------
template <int ACT>
__global__ void gemm_bf16_wmma_tdm(const __bf16* __restrict__ A, int lda,
                                   const __bf16* __restrict__ W, int ldw,
                                   const float* __restrict__ bias,
                                   float* __restrict__ C, int ldc,
                                   int M, int N, int K) {
  __shared__ __bf16 ldsA[2][64 * 32];
  __shared__ __bf16 ldsB[2][64 * 32];
  const int lane = threadIdx.x & 31;
  const int wave = threadIdx.x >> 5;     // 0..7
  const int wm   = wave >> 1;            // 0..3 (M)
  const int wn   = wave & 1;             // 0..1 (N)
  const int rowB = blockIdx.y * 64;
  const int colB = blockIdx.x * 64;
  const int colW = wn * 32;              // wave N offset inside LDS tile

  const int nsteps = K >> 5;

  if (wave == 0) {
    tdm_load_2d_bf16(A + (size_t)rowB * lda, (unsigned)(uintptr_t)&ldsA[0][0],
                     (unsigned)lda, (unsigned)M, 32u, 64u, (unsigned)lda);
    tdm_load_2d_bf16(W + (size_t)colB * ldw, (unsigned)(uintptr_t)&ldsB[0][0],
                     (unsigned)ldw, (unsigned)N, 32u, 64u, (unsigned)ldw);
  }

  v8f c0 = {};
  v8f c1 = {};

  for (int i = 0; i < nsteps; ++i) {
    const int buf = i & 1;
    if (wave == 0) {
      if (i + 1 < nsteps) {
        const int k1 = (i + 1) << 5;
        tdm_load_2d_bf16(A + (size_t)rowB * lda + k1,
                         (unsigned)(uintptr_t)&ldsA[buf ^ 1][0],
                         (unsigned)lda, (unsigned)M, 32u, 64u, (unsigned)lda);
        tdm_load_2d_bf16(W + (size_t)colB * ldw + k1,
                         (unsigned)(uintptr_t)&ldsB[buf ^ 1][0],
                         (unsigned)ldw, (unsigned)N, 32u, 64u, (unsigned)ldw);
        __builtin_amdgcn_s_wait_tensorcnt(2);   // current stage resident
      } else {
        __builtin_amdgcn_s_wait_tensorcnt(0);   // final stage resident
      }
    }
    __syncthreads();                            // stage visible to all waves

    // A fragment from LDS (row-major 64x32 tile, 64B rows)
    const int m  = lane & 15;
    const int kh = (lane >> 4) << 3;
    const __bf16* pa = &ldsA[buf][(wm * 16 + m) * 32 + kh];
    v16bf a = cat16(*(const v8bf*)pa, *(const v8bf*)(pa + 16));

    // B fragments from LDS
    const int n  = lane & 15;
    const int kb = (lane >> 4) << 3;
    const __bf16* q0 = &ldsB[buf][(colW + n) * 32 + kb];
    const __bf16* q1 = &ldsB[buf][(colW + 16 + n) * 32 + kb];
    v16bf b0 = cat16(*(const v8bf*)q0, *(const v8bf*)(q0 + 16));
    v16bf b1 = cat16(*(const v8bf*)q1, *(const v8bf*)(q1 + 16));

    c0 = __builtin_amdgcn_wmma_f32_16x16x32_bf16(false, a, false, b0,
                                                 (short)0, c0, false, false);
    c1 = __builtin_amdgcn_wmma_f32_16x16x32_bf16(false, a, false, b1,
                                                 (short)0, c1, false, false);
    __syncthreads();                            // reads done before rewrite
  }

  const int n    = lane & 15;
  const int mofs = (lane >> 4) << 3;
  const int row0 = rowB + wm * 16;
  const int col0 = colB + colW;
  const float bia0 = bias ? bias[col0 + n]      : 0.0f;
  const float bia1 = bias ? bias[col0 + 16 + n] : 0.0f;
#pragma unroll
  for (int r = 0; r < 8; ++r) {
    const int row = row0 + mofs + r;
    float v0 = c0[r] + bia0;
    float v1 = c1[r] + bia1;
    if (ACT == 1) { v0 = softplus_f(v0); v1 = softplus_f(v1); }
    C[(size_t)row * ldc + col0 + n]      = v0;
    C[(size_t)row * ldc + col0 + 16 + n] = v1;
  }
}

// ---------------------------------------------------------------------------
// Small GEMM (N=96 path): direct bf16 global loads, one wave = 16x32 tile.
// ---------------------------------------------------------------------------
__global__ void gemm_bf16_wmma_direct(const __bf16* __restrict__ A, int lda,
                                      const __bf16* __restrict__ W, int ldw,
                                      float* __restrict__ C, int ldc,
                                      int M, int N, int K) {
  const int lane = threadIdx.x & 31;
  const int wave = threadIdx.x >> 5;
  const int gw   = blockIdx.x * (blockDim.x >> 5) + wave;
  const int nt2  = N >> 5;
  const int mt   = gw / nt2;
  const int nt   = gw % nt2;
  if (mt * 16 >= M) return;
  const int row0 = mt * 16;
  const int col0 = nt * 32;

  v8f c0 = {};
  v8f c1 = {};
  for (int k0 = 0; k0 < K; k0 += 32) {
    if (k0 + 32 < K)
      __builtin_prefetch(A + (size_t)(row0 + (lane & 15)) * lda + k0 + 32, 0, 1);
    v16bf a  = load_A_frag_bf(A, lda, row0, k0, lane);
    v16bf b0 = load_B_frag_bf(W, ldw, col0,      k0, lane);
    v16bf b1 = load_B_frag_bf(W, ldw, col0 + 16, k0, lane);
    c0 = __builtin_amdgcn_wmma_f32_16x16x32_bf16(false, a, false, b0,
                                                 (short)0, c0, false, false);
    c1 = __builtin_amdgcn_wmma_f32_16x16x32_bf16(false, a, false, b1,
                                                 (short)0, c1, false, false);
  }
  const int n    = lane & 15;
  const int mofs = (lane >> 4) << 3;
#pragma unroll
  for (int r = 0; r < 8; ++r) {
    const int row = row0 + mofs + r;
    C[(size_t)row * ldc + col0 + n]      = c0[r];
    C[(size_t)row * ldc + col0 + 16 + n] = c1[r];
  }
}

// ---------------------------------------------------------------------------
// fp32 -> bf16 conversion (n must be a multiple of 4)
// ---------------------------------------------------------------------------
__global__ void cvt_f32_bf16(const float* __restrict__ in, __bf16* __restrict__ out,
                             int n) {
  const int i = (blockIdx.x * blockDim.x + threadIdx.x) * 4;
  if (i >= n) return;
  const float4 v = *(const float4*)(in + i);
  out[i + 0] = (__bf16)v.x;
  out[i + 1] = (__bf16)v.y;
  out[i + 2] = (__bf16)v.z;
  out[i + 3] = (__bf16)v.w;
}

// ---------------------------------------------------------------------------
// Causal depthwise conv (k=4) + bias + SiLU; writes fp32 (scan) + bf16 (GEMM).
// ---------------------------------------------------------------------------
__global__ void conv_silu_kernel(const float* __restrict__ xz,
                                 const float* __restrict__ cw,
                                 const float* __restrict__ cb,
                                 float* __restrict__ xc,
                                 __bf16* __restrict__ xc_bf) {
  const int idx = blockIdx.x * blockDim.x + threadIdx.x;
  if (idx >= BB * LL * D_INNER) return;
  const int d = idx % D_INNER;
  const int l = (idx / D_INNER) % LL;
  const int b = idx / (D_INNER * LL);

  const float* xi = xz + (size_t)b * LL * (2 * D_INNER);
  float acc = cb[d];
#pragma unroll
  for (int i = 0; i < D_CONV; ++i) {
    const int ls = l - (D_CONV - 1) + i;
    if (ls >= 0) acc += xi[(size_t)ls * (2 * D_INNER) + d] * cw[d * D_CONV + i];
  }
  const float s = silu_f(acc);
  xc[idx]    = s;
  xc_bf[idx] = (__bf16)s;
}

// ---------------------------------------------------------------------------
// Selective scan: one thread per (b,d) channel, h[16] in registers.
// Fuses y = (sum_n h*C + xc*D) * silu(z); writes bf16 for the out-projection.
// ---------------------------------------------------------------------------
__global__ void selective_scan_kernel(const float* __restrict__ delta,
                                      const float* __restrict__ xc,
                                      const float* __restrict__ xdbl,
                                      const float* __restrict__ xz,
                                      const float* __restrict__ A_log,
                                      const float* __restrict__ Dp,
                                      __bf16* __restrict__ y_bf) {
  const int tid = blockIdx.x * blockDim.x + threadIdx.x;
  if (tid >= BB * D_INNER) return;
  const int b = tid / D_INNER;
  const int d = tid % D_INNER;

  float Arow[D_STATE];
#pragma unroll
  for (int n = 0; n < D_STATE; ++n) Arow[n] = -__expf(A_log[d * D_STATE + n]);
  const float ddp = Dp[d];

  float h[D_STATE];
#pragma unroll
  for (int n = 0; n < D_STATE; ++n) h[n] = 0.0f;

  for (int l = 0; l < LL; ++l) {
    const size_t tok = (size_t)b * LL + l;
    const float dv  = delta[tok * D_INNER + d];
    const float xv  = xc[tok * D_INNER + d];
    const float* bc = xdbl + tok * (DT_RANK + 2 * D_STATE);
    const float dxv = dv * xv;

    float acc = 0.0f;
#pragma unroll
    for (int n = 0; n < D_STATE; ++n) {
      const float dA = __expf(dv * Arow[n]);
      h[n] = dA * h[n] + dxv * bc[DT_RANK + n];
      acc += h[n] * bc[DT_RANK + D_STATE + n];
    }
    const float zf = xz[tok * (2 * D_INNER) + D_INNER + d];
    y_bf[tok * D_INNER + d] = (__bf16)((acc + xv * ddp) * silu_f(zf));
  }
}

// ---------------------------------------------------------------------------
// Host launcher
// ---------------------------------------------------------------------------
extern "C" void kernel_launch(void* const* d_in, const int* in_sizes, int n_in,
                              void* d_out, int out_size, void* d_ws, size_t ws_size,
                              hipStream_t stream) {
  const float* x          = (const float*)d_in[0];  // (B,L,1024)
  const float* in_proj_w  = (const float*)d_in[1];  // (4096,1024)
  const float* conv_w     = (const float*)d_in[2];  // (2048,4)
  const float* conv_b     = (const float*)d_in[3];  // (2048)
  const float* x_proj_w   = (const float*)d_in[4];  // (96,2048)
  const float* dt_proj_w  = (const float*)d_in[5];  // (2048,64)
  const float* dt_proj_b  = (const float*)d_in[6];  // (2048)
  const float* A_log      = (const float*)d_in[7];  // (2048,16)
  const float* Dp         = (const float*)d_in[8];  // (2048)
  const float* out_proj_w = (const float*)d_in[9];  // (1024,2048)
  float* out = (float*)d_out;                       // (B,L,1024)

  // ---- workspace layout ----
  float* ws    = (float*)d_ws;
  float* xz    = ws;                                              // 2048*4096
  float* xc    = xz    + (size_t)M_TOK * 2 * D_INNER;             // 2048*2048
  float* xdbl  = xc    + (size_t)M_TOK * D_INNER;                 // 2048*96
  float* delta = xdbl  + (size_t)M_TOK * (DT_RANK + 2 * D_STATE); // 2048*2048
  __bf16* bfbase   = (__bf16*)(delta + (size_t)M_TOK * D_INNER);
  __bf16* x_bf     = bfbase;                                      // 2048*1024
  __bf16* inw_bf   = x_bf     + (size_t)M_TOK * D_MODEL;          // 4096*1024
  __bf16* xprojw_bf= inw_bf   + (size_t)2 * D_INNER * D_MODEL;    // 96*2048
  __bf16* dtw_bf   = xprojw_bf+ (size_t)(DT_RANK + 2*D_STATE) * D_INNER; // 2048*64
  __bf16* outw_bf  = dtw_bf   + (size_t)D_INNER * DT_RANK;        // 1024*2048
  __bf16* xc_bf    = outw_bf  + (size_t)D_MODEL * D_INNER;        // 2048*2048
  __bf16* xdbl_bf  = xc_bf    + (size_t)M_TOK * D_INNER;          // 2048*96
  __bf16* y_bf     = xdbl_bf  + (size_t)M_TOK * (DT_RANK + 2*D_STATE); // 2048*2048

  auto cvt = [&](const float* src, __bf16* dst, size_t n) {
    cvt_f32_bf16<<<dim3((unsigned)((n / 4 + 255) / 256)), dim3(256), 0, stream>>>(
        src, dst, (int)n);
  };

  // 0) bf16 copies of all GEMM operands (weights + x)
  cvt(x,          x_bf,      (size_t)M_TOK * D_MODEL);
  cvt(in_proj_w,  inw_bf,    (size_t)2 * D_INNER * D_MODEL);
  cvt(x_proj_w,   xprojw_bf, (size_t)(DT_RANK + 2 * D_STATE) * D_INNER);
  cvt(dt_proj_w,  dtw_bf,    (size_t)D_INNER * DT_RANK);
  cvt(out_proj_w, outw_bf,   (size_t)D_MODEL * D_INNER);

  // 1) xz = x @ in_proj_w^T     M=2048 N=4096 K=1024  (TDM A+B, double-buffered)
  gemm_bf16_wmma_tdm<0><<<dim3((2 * D_INNER) / 64, M_TOK / 64), dim3(256), 0, stream>>>(
      x_bf, D_MODEL, inw_bf, D_MODEL, nullptr, xz, 2 * D_INNER,
      M_TOK, 2 * D_INNER, D_MODEL);

  // 2) xc = silu(causal_conv(xi) + b)
  {
    int n = BB * LL * D_INNER;
    conv_silu_kernel<<<dim3((n + 255) / 256), dim3(256), 0, stream>>>(
        xz, conv_w, conv_b, xc, xc_bf);
  }

  // 3) x_dbl = xc @ x_proj_w^T   M=2048 N=96 K=2048   (direct path, N%64!=0)
  {
    int Nn = DT_RANK + 2 * D_STATE;  // 96
    int waves = (M_TOK / 16) * (Nn / 32);
    gemm_bf16_wmma_direct<<<dim3(waves / 4), dim3(128), 0, stream>>>(
        xc_bf, D_INNER, xprojw_bf, D_INNER, xdbl, Nn, M_TOK, Nn, D_INNER);
  }

  // 3b) bf16 copy of x_dbl (dt columns feed the delta GEMM)
  cvt(xdbl, xdbl_bf, (size_t)M_TOK * (DT_RANK + 2 * D_STATE));

  // 4) delta = softplus(dt @ dt_proj_w^T + b)  M=2048 N=2048 K=64 (TDM)
  gemm_bf16_wmma_tdm<1><<<dim3(D_INNER / 64, M_TOK / 64), dim3(256), 0, stream>>>(
      xdbl_bf, DT_RANK + 2 * D_STATE, dtw_bf, DT_RANK, dt_proj_b,
      delta, D_INNER, M_TOK, D_INNER, DT_RANK);

  // 5) selective scan + gating -> y (bf16)
  selective_scan_kernel<<<dim3((BB * D_INNER) / 256), dim3(256), 0, stream>>>(
      delta, xc, xdbl, xz, A_log, Dp, y_bf);

  // 6) out = y @ out_proj_w^T    M=2048 N=1024 K=2048 (TDM)
  gemm_bf16_wmma_tdm<0><<<dim3(D_MODEL / 64, M_TOK / 64), dim3(256), 0, stream>>>(
      y_bf, D_INNER, outw_bf, D_INNER, nullptr, out, D_MODEL,
      M_TOK, D_MODEL, D_INNER);
}
